// PenalizedAttention_77575699300582
// MI455X (gfx1250) — matrix operand order repaired and checked by reading, not compiled, not run on hardware
//
#include <hip/hip_runtime.h>

// ---------------------------------------------------------------------------
// Penalized attention for MI455X (gfx1250, wave32, WMMA).
// softmax(P - revcumsum(sig(P)_masked)) == softmax(P + excl_prefix(sig(P)_masked))
// -> single-pass flash-style attention with an online prefix carry.
// All GEMMs use v_wmma_f32_16x16x32_bf16 (bf16 in, f32 accumulate).
// Scans/reductions use DPP16 (row_shr / row_ror) -> pure VALU, no ds_bpermute.
// ---------------------------------------------------------------------------

namespace pa {
constexpr int  Bn = 2, Tn = 2048, Cn = 1024, Hn = 16, Dn = 64;
constexpr float kScale = 0.125f; // 1/sqrt(64)
}

typedef __attribute__((ext_vector_type(16))) __bf16 v16bf;
typedef __attribute__((ext_vector_type(8)))  __bf16 v8bf;
typedef __attribute__((ext_vector_type(8)))  float  v8f;

__device__ __forceinline__ unsigned short f2bf(float f) {
  union { float f; unsigned int u; } c; c.f = f;
  unsigned int u = c.u;
  u += 0x7FFFu + ((u >> 16) & 1u);   // round-to-nearest-even
  return (unsigned short)(u >> 16);
}

__device__ __forceinline__ float fast_rcp(float x) {
  return __builtin_amdgcn_rcpf(x);   // single v_rcp_f32
}

// ---- DPP16 helpers (rows of 16 lanes == our scan groups) -------------------
template <int N>  // row_shr:N, zero-fill at row edge
__device__ __forceinline__ float dpp_sr0(float x) {
  return __builtin_bit_cast(float, __builtin_amdgcn_update_dpp(
      0, __builtin_bit_cast(int, x), 0x110 | N, 0xF, 0xF, true));
}
template <int N>  // row_ror:N (full rotation, all lanes valid)
__device__ __forceinline__ float dpp_ror(float x) {
  return __builtin_bit_cast(float, __builtin_amdgcn_update_dpp(
      0, __builtin_bit_cast(int, x), 0x120 | N, 0xF, 0xF, true));
}
__device__ __forceinline__ float scan_incl16(float v) {  // inclusive prefix
  v += dpp_sr0<1>(v);
  v += dpp_sr0<2>(v);
  v += dpp_sr0<4>(v);
  v += dpp_sr0<8>(v);
  return v;
}
__device__ __forceinline__ float sum16(float v) {        // all lanes get total
  v += dpp_ror<8>(v);
  v += dpp_ror<4>(v);
  v += dpp_ror<2>(v);
  v += dpp_ror<1>(v);
  return v;
}
__device__ __forceinline__ float max16(float v) {        // all lanes get max
  v = fmaxf(v, dpp_ror<8>(v));
  v = fmaxf(v, dpp_ror<4>(v));
  v = fmaxf(v, dpp_ror<2>(v));
  v = fmaxf(v, dpp_ror<1>(v));
  return v;
}

// A-operand fragment (16x32 bf16): lane = row (lane&15); per lane the K
// elements are 16 contiguous halfs at base = grp*16.
__device__ __forceinline__ v16bf frag_a(const unsigned short* p) {
  union { v16bf v; v8bf h[2]; } u;
  u.h[0] = *reinterpret_cast<const v8bf*>(p);
  u.h[1] = *reinterpret_cast<const v8bf*>(p + 8);
  return u.v;
}
// B-operand fragment (32x16 bf16): lane = col (lane&15); K elements are 16
// contiguous halfs at base = grp*16.
__device__ __forceinline__ v16bf frag_b(const unsigned short* p) {
  return *reinterpret_cast<const v16bf*>(p);
}

__device__ __forceinline__ v8f wmma_bf16(v16bf a, v16bf b, v8f c) {
  return __builtin_amdgcn_wmma_f32_16x16x32_bf16(false, a, false, b,
                                                 (short)0, c, false, false);
}

// ---------------------------------------------------------------------------
__global__ void pa_cvt_f32_bf16(const float* __restrict__ in,
                                unsigned short* __restrict__ out, int n) {
  for (int i = blockIdx.x * blockDim.x + threadIdx.x; i < n;
       i += gridDim.x * blockDim.x)
    out[i] = f2bf(in[i]);
}

// qkv[B*T, 3C] -> q[BH,T,D], k[BH,T,D], vT[BH,D,T]
__global__ void pa_repack_qkv(const unsigned short* __restrict__ qkv,
                              unsigned short* __restrict__ q,
                              unsigned short* __restrict__ k,
                              unsigned short* __restrict__ vT) {
  using namespace pa;
  const int total = Bn * Tn * 3 * Cn;
  for (int i = blockIdx.x * blockDim.x + threadIdx.x; i < total;
       i += gridDim.x * blockDim.x) {
    int m = i / (3 * Cn), n = i % (3 * Cn);
    int b = m / Tn, t = m % Tn;
    int which = n / Cn, cc = n % Cn;
    int h = cc / Dn, d = cc % Dn;
    unsigned short v = qkv[i];
    size_t bh = (size_t)b * Hn + h;
    if (which == 0)      q[(bh * Tn + t) * Dn + d] = v;
    else if (which == 1) k[(bh * Tn + t) * Dn + d] = v;
    else                 vT[(bh * Dn + d) * Tn + t] = v;
  }
}

// ---------------------------------------------------------------------------
// C[M,N] = A[M,K] * B[N,K]^T  (bf16 in, f32 accumulate).
// 256 threads = 8 waves; wave tile 32x32 (2x2 WMMA tiles); block tile 128x64.
template <bool OUTF32>
__global__ __launch_bounds__(256) void pa_gemm_nt(
    const unsigned short* __restrict__ A, const unsigned short* __restrict__ Bm,
    float* __restrict__ Cf, unsigned short* __restrict__ Cb,
    int M, int N, int K) {
  const int wave = threadIdx.x >> 5, lane = threadIdx.x & 31;
  const int grp = lane >> 4, ln = lane & 15;
  const int mw = wave & 3, nw = wave >> 2;
  const int m0 = blockIdx.y * 128 + mw * 32;
  const int n0 = blockIdx.x * 64 + nw * 32;

  const unsigned short* a0p = A + (size_t)(m0 + ln) * K;
  const unsigned short* a1p = A + (size_t)(m0 + 16 + ln) * K;
  const unsigned short* b0p = Bm + (size_t)(n0 + ln) * K;
  const unsigned short* b1p = Bm + (size_t)(n0 + 16 + ln) * K;

  v8f acc00 = {}, acc01 = {}, acc10 = {}, acc11 = {};
  for (int k0 = 0; k0 < K; k0 += 32) {
    if (k0 + 256 < K) {  // L2 warm-up for the next stretch of K
      __builtin_prefetch(a0p + k0 + 256, 0, 0);
      __builtin_prefetch(b0p + k0 + 256, 0, 0);
    }
    v16bf a0 = frag_a(a0p + k0 + grp * 16);
    v16bf a1 = frag_a(a1p + k0 + grp * 16);
    v16bf b0 = frag_b(b0p + k0 + grp * 16);
    v16bf b1 = frag_b(b1p + k0 + grp * 16);
    acc00 = wmma_bf16(a0, b0, acc00);
    acc01 = wmma_bf16(a0, b1, acc01);
    acc10 = wmma_bf16(a1, b0, acc10);
    acc11 = wmma_bf16(a1, b1, acc11);
  }

  // C/D layout: VGPR r of lane -> row r + 8*grp, col = ln (per 16x16 tile).
#pragma unroll
  for (int ti = 0; ti < 2; ++ti)
#pragma unroll
    for (int tj = 0; tj < 2; ++tj) {
      const v8f& acc = ti == 0 ? (tj == 0 ? acc00 : acc01)
                               : (tj == 0 ? acc10 : acc11);
#pragma unroll
      for (int r = 0; r < 8; ++r) {
        int row = m0 + ti * 16 + r + 8 * grp;
        int col = n0 + tj * 16 + ln;
        if (OUTF32) Cf[(size_t)row * N + col] = acc[r];
        else        Cb[(size_t)row * N + col] = f2bf(acc[r]);
      }
    }
}

// ---------------------------------------------------------------------------
// Flash-style penalized attention. 4 waves/block, 16 queries/wave.
__global__ __launch_bounds__(128) void pa_attn(
    const unsigned short* __restrict__ Q, const unsigned short* __restrict__ Kt,
    const unsigned short* __restrict__ Vt, unsigned short* __restrict__ Y) {
  using namespace pa;
  __shared__ unsigned short smem[4][16 * 40];  // 16 rows x 32 kk, stride 40

  const int bh = blockIdx.x;                   // b*H + h
  const int wave = threadIdx.x >> 5, lane = threadIdx.x & 31;
  const int grp = lane >> 4, ln = lane & 15;
  const int qBase = blockIdx.y * 64 + wave * 16;
  const int b = bh / Hn, h = bh % Hn;

  const unsigned short* qp = Q + ((size_t)bh * Tn + qBase) * Dn;
  const unsigned short* kp = Kt + (size_t)bh * Tn * Dn;
  const unsigned short* vp = Vt + (size_t)bh * Dn * Tn;

  // Query A-fragments (K = d, 0..63 split into two K=32 frags).
  const unsigned short* qrow = qp + (size_t)ln * Dn;
  v16bf aq0 = frag_a(qrow + grp * 16);
  v16bf aq1 = frag_a(qrow + 32 + grp * 16);

  float m_run[8], d_run[8], pref[8];
  v8f yacc[4];
#pragma unroll
  for (int r = 0; r < 8; ++r) { m_run[r] = -3.0e38f; d_run[r] = 0.f; pref[r] = 0.f; }
#pragma unroll
  for (int dt = 0; dt < 4; ++dt) yacc[dt] = (v8f){};

  const float NEG = -3.0e38f;
  const int kEnd = qBase + 16;
  for (int kb = 0; kb < kEnd; kb += 32) {
    // ---- P = (q . k) * scale for keys [kb, kb+32) -------------------------
    const unsigned short* k0r = kp + (size_t)(kb + ln) * Dn;
    const unsigned short* k1r = kp + (size_t)(kb + 16 + ln) * Dn;
    v8f p0 = {}, p1 = {};
    p0 = wmma_bf16(aq0, frag_b(k0r + grp * 16), p0);
    p0 = wmma_bf16(aq1, frag_b(k0r + 32 + grp * 16), p0);
    p1 = wmma_bf16(aq0, frag_b(k1r + grp * 16), p1);
    p1 = wmma_bf16(aq1, frag_b(k1r + 32 + grp * 16), p1);

    // ---- per-row: sigmoid prefix, online softmax (all DPP, no LDS) --------
#pragma unroll
    for (int r = 0; r < 8; ++r) {
      int row = r + 8 * grp;
      int qg = qBase + row;
      int kg0 = kb + ln, kg1 = kb + 16 + ln;
      bool a0 = kg0 <= qg, a1 = kg1 <= qg;

      float P0 = p0[r] * kScale, P1 = p1[r] * kScale;
      // sigmoid via single v_rcp_f32 (branch-free, no IEEE divide chain)
      float g0 = fast_rcp(1.0f + __expf(-P0));
      float g1 = fast_rcp(1.0f + __expf(-P1));
      float s0 = a0 ? g0 : 0.f;
      float s1 = a1 ? g1 : 0.f;

      // inclusive 16-lane scans along keys + per-tile totals (DPP16)
      float inc0 = scan_incl16(s0);
      float ex0  = inc0 - s0;
      float sum0 = sum16(s0);
      float inc1 = scan_incl16(s1);
      float ex1  = inc1 - s1;
      float sum1 = sum16(s1);

      float L0 = P0 + pref[r] + ex0;
      float L1 = P1 + pref[r] + sum0 + ex1;
      pref[r] += sum0 + sum1;

      float mb = max16(fmaxf(a0 ? L0 : NEG, a1 ? L1 : NEG));
      float mn = fmaxf(m_run[r], mb);
      float alpha = __expf(m_run[r] - mn);
      float e0 = a0 ? __expf(L0 - mn) : 0.f;
      float e1 = a1 ? __expf(L1 - mn) : 0.f;
      float rs = sum16(e0 + e1);
      d_run[r] = d_run[r] * alpha + rs;
      m_run[r] = mn;
#pragma unroll
      for (int dt = 0; dt < 4; ++dt) yacc[dt][r] *= alpha;

      // intra-wave LDS transpose (C-layout -> A-layout); same-wave DS ops
      // are in order, so no barrier needed.
      smem[wave][row * 40 + ln] = f2bf(e0);
      smem[wave][row * 40 + 16 + ln] = f2bf(e1);
    }

    // ---- y += p[16x32] . v[32x16] per 16-wide d tile ----------------------
    v16bf ap = frag_a(&smem[wave][ln * 40 + grp * 16]);
#pragma unroll
    for (int dt = 0; dt < 4; ++dt) {
      const unsigned short* vrow = vp + (size_t)(dt * 16 + ln) * Tn;
      yacc[dt] = wmma_bf16(ap, frag_b(vrow + kb + grp * 16), yacc[dt]);
    }
  }

  // normalize (one v_rcp per row) and scatter to y[B,T,C]
  float inv_d[8];
#pragma unroll
  for (int r = 0; r < 8; ++r) inv_d[r] = fast_rcp(d_run[r]);
#pragma unroll
  for (int dt = 0; dt < 4; ++dt)
#pragma unroll
    for (int r = 0; r < 8; ++r) {
      int row = r + 8 * grp;
      int qg = qBase + row;
      float val = yacc[dt][r] * inv_d[r];
      Y[((size_t)b * Tn + qg) * Cn + (size_t)h * Dn + dt * 16 + ln] = f2bf(val);
    }
}

// ---------------------------------------------------------------------------
extern "C" void kernel_launch(void* const* d_in, const int* in_sizes, int n_in,
                              void* d_out, int out_size, void* d_ws, size_t ws_size,
                              hipStream_t stream) {
  using namespace pa;
  (void)in_sizes; (void)n_in; (void)out_size; (void)ws_size;
  const float* x  = (const float*)d_in[0];
  const float* Wa = (const float*)d_in[1];
  const float* Wp = (const float*)d_in[2];
  float* out = (float*)d_out;

  char* ws = (char*)d_ws;
  size_t off = 0;
  auto give = [&](size_t elems) -> unsigned short* {
    unsigned short* p = (unsigned short*)(ws + off);
    off += (elems * 2 + 255) & ~(size_t)255;
    return p;
  };
  unsigned short* xbf  = give((size_t)Bn * Tn * Cn);        // 8 MB
  unsigned short* wabf = give((size_t)3 * Cn * Cn);         // 6 MB
  unsigned short* wpbf = give((size_t)Cn * Cn);             // 2 MB
  unsigned short* qkv  = give((size_t)Bn * Tn * 3 * Cn);    // 24 MB
  unsigned short* qb   = give((size_t)Bn * Hn * Tn * Dn);   // 8 MB
  unsigned short* kb   = give((size_t)Bn * Hn * Tn * Dn);   // 8 MB
  unsigned short* vT   = give((size_t)Bn * Hn * Dn * Tn);   // 8 MB
  unsigned short* yb   = give((size_t)Bn * Tn * Cn);        // 8 MB

  pa_cvt_f32_bf16<<<1024, 256, 0, stream>>>(x, xbf, Bn * Tn * Cn);
  pa_cvt_f32_bf16<<<1024, 256, 0, stream>>>(Wa, wabf, 3 * Cn * Cn);
  pa_cvt_f32_bf16<<<512, 256, 0, stream>>>(Wp, wpbf, Cn * Cn);

  // qkv = x @ W_attn^T : M=4096, N=3072, K=1024
  pa_gemm_nt<false><<<dim3(3 * Cn / 64, Bn * Tn / 128), 256, 0, stream>>>(
      xbf, wabf, nullptr, qkv, Bn * Tn, 3 * Cn, Cn);

  pa_repack_qkv<<<2048, 256, 0, stream>>>(qkv, qb, kb, vT);

  // flash penalized attention: one wave per 16 queries
  pa_attn<<<dim3(Bn * Hn, Tn / 64), 128, 0, stream>>>(qb, kb, vT, yb);

  // out = y @ W_proj^T : M=4096, N=1024, K=1024 (f32 output)
  pa_gemm_nt<true><<<dim3(Cn / 64, Bn * Tn / 128), 256, 0, stream>>>(
      yb, wpbf, out, nullptr, Bn * Tn, Cn, Cn);
}
